// Im2LatexModel_6330781794326
// MI455X (gfx1250) — matrix operand order repaired and checked by reading, not compiled, not run on hardware
//
#include <hip/hip_runtime.h>

// ---------------------------------------------------------------------------
// Model dims
// ---------------------------------------------------------------------------
#define Bm_     16
#define Lm_     512
#define Dm_     512
#define NH_     8
#define HD_     64
#define LK_     49
#define DI_     1024
#define DS_     16
#define DTR_    32
#define FFN_    2048
#define VOC_    8000
#define ROWS_   (Bm_ * Lm_)      // 8192
#define MROWS_  (Bm_ * LK_)      // 784

typedef __attribute__((ext_vector_type(16))) _Float16 v16h;
typedef __attribute__((ext_vector_type(8)))  _Float16 v8h;
typedef __attribute__((ext_vector_type(8)))  float    v8f;

#define ACT_NONE     0
#define ACT_GELU     1
#define ACT_SOFTPLUS 2

__device__ __forceinline__ float gelu_f(float x) {
    float x3 = x * x * x;
    return 0.5f * x * (1.f + tanhf(0.7978845608028654f * (x + 0.044715f * x3)));
}
__device__ __forceinline__ float softplus_f(float x) {
    return (x > 20.f) ? x : log1pf(__expf(x));
}
__device__ __forceinline__ float silu_f(float x) {
    return x / (1.f + __expf(-x));
}

// A-fragment loader: 16x32 f16 tile row for one lane (contiguous K order):
// lane<16 -> K {0..15}; lane>=16 -> K {16..31}; kbaseA = 0 or 8.
__device__ __forceinline__ v16h load_a_frag(const _Float16* __restrict__ Arow,
                                            int k0, int kbaseA)
{
    v8h lo = *(const v8h*)(Arow + k0 + 2 * kbaseA);
    v8h hi = *(const v8h*)(Arow + k0 + 8 + 2 * kbaseA);
    v16h a;
#pragma unroll
    for (int i = 0; i < 8; ++i) { a[i] = lo[i]; a[8 + i] = hi[i]; }
    return a;
}

// ---------------------------------------------------------------------------
// LDS-staged WMMA GEMM:  C[M,N] = act(A[M,K(lda)] * W[N,K]^T + bias[N])
// Requirements: N % 64 == 0, K % 64 == 0, M % 16 == 0.
// Workgroup = 8 waves stacked along M; each wave computes 32(M) x 64(N).
// The 64 weight rows of the workgroup's N strip are staged once per K-chunk
// in LDS and shared by all 8 waves (8x cut in global/L2 B traffic).
// Single-LDS-buffer pipeline: next chunk is prefetched into registers while
// the current chunk is consumed between the two barriers.
// ---------------------------------------------------------------------------
__global__ __launch_bounds__(256)
void gemm_wmma_f16_lds(const _Float16* __restrict__ A, int lda,
                       const _Float16* __restrict__ W,       // [N,K] row-major
                       const float* __restrict__ bias,
                       float* __restrict__ C, int ldc,
                       int M, int N, int K, int act)
{
    constexpr int BK   = 64;   // K-chunk (two WMMA K-steps)
    constexpr int SROW = 80;   // padded LDS row stride in halfs (160B: 32B-aligned,
                               // bank-staggered: 40 dwords/row)
    __shared__ __align__(32) _Float16 sB[64 * SROW];

    const int tid   = threadIdx.x;
    const int wave  = tid >> 5;
    const int lane  = tid & 31;
    const int mtile = (blockIdx.y * 8 + wave) * 32;
    const int ntile = blockIdx.x * 64;

    const bool v0 = (mtile     ) < M;    // first 16-row frag valid
    const bool v1 = (mtile + 16) < M;    // second 16-row frag valid

    const int kbaseA = (lane >> 4) * 8;    // 0 or 8
    const int kbaseB = (lane >> 4) * 16;   // 0 or 16
    const int ncol   = lane & 15;

    const _Float16* Arow0 = A + (size_t)((v0 ? mtile : 0) + (lane & 15)) * lda;
    const _Float16* Arow1 = v1 ? (Arow0 + (size_t)16 * lda) : Arow0;

    // Cooperative fill assignment: thread -> (weight row, 16-half segment)
    const int frow = tid >> 2;            // 0..63
    const int fseg = (tid & 3) * 16;      // 0,16,32,48
    const _Float16* Wfill = W + (size_t)(ntile + frow) * K + fseg;

    v8f acc[2][4];
#pragma unroll
    for (int f = 0; f < 2; ++f)
#pragma unroll
        for (int j = 0; j < 4; ++j) acc[f][j] = v8f{};

    v16h wreg = *(const v16h*)(Wfill);    // chunk 0 in registers
    for (int k0 = 0; k0 < K; k0 += BK) {
        *(v16h*)(&sB[frow * SROW + fseg]) = wreg;
        __syncthreads();
        if (k0 + BK < K) {
            wreg = *(const v16h*)(Wfill + k0 + BK);        // prefetch next chunk
            __builtin_prefetch(Arow0 + k0 + BK + kbaseA, 0, 1);
        }
#pragma unroll
        for (int ks = 0; ks < BK; ks += 32) {
            v16h a0 = load_a_frag(Arow0, k0 + ks, kbaseA);
            v16h a1 = load_a_frag(Arow1, k0 + ks, kbaseA);
#pragma unroll
            for (int j = 0; j < 4; ++j) {
                v16h b = *(const v16h*)(&sB[(j * 16 + ncol) * SROW + ks + kbaseB]);
                acc[0][j] = __builtin_amdgcn_wmma_f32_16x16x32_f16(
                    false, a0, false, b, (short)0, acc[0][j], false, false);
                acc[1][j] = __builtin_amdgcn_wmma_f32_16x16x32_f16(
                    false, a1, false, b, (short)0, acc[1][j], false, false);
            }
        }
        __syncthreads();
    }

    // C layout: VGPR r holds row (tile + r + 8*(lane>=16)), column ntile+j*16+ncol
    const int mbase = mtile + ((lane >> 4) * 8);
#pragma unroll
    for (int f = 0; f < 2; ++f) {
        if (f == 0 ? !v0 : !v1) continue;
#pragma unroll
        for (int j = 0; j < 4; ++j) {
            int n = ntile + j * 16 + ncol;
            float bs = bias ? bias[n] : 0.f;
#pragma unroll
            for (int r = 0; r < 8; ++r) {
                float v = acc[f][j][r] + bs;
                if (act == ACT_GELU)          v = gelu_f(v);
                else if (act == ACT_SOFTPLUS) v = softplus_f(v);
                C[(size_t)(mbase + f * 16 + r) * ldc + n] = v;
            }
        }
    }
}

// ---------------------------------------------------------------------------
// Unstaged WMMA GEMM (fallback: K%64!=0 or N%64!=0)
// ---------------------------------------------------------------------------
template<int MFRAG, bool FULLN>
__global__ __launch_bounds__(256)
void gemm_wmma_f16(const _Float16* __restrict__ A, int lda,
                   const _Float16* __restrict__ W,           // [N,K] row-major
                   const float* __restrict__ bias,
                   float* __restrict__ C, int ldc,
                   int M, int N, int K, int act)
{
    const int wave  = threadIdx.x >> 5;
    const int lane  = threadIdx.x & 31;
    const int mtile = (blockIdx.y * 8 + wave) * (16 * MFRAG);
    if (mtile >= M) return;
    const int ntile = blockIdx.x * 64;

    const int kbaseA = (lane >> 4) * 8;
    const int kbaseB = (lane >> 4) * 16;
    const int ncol   = lane & 15;

    const _Float16* Arow0 = A + (size_t)(mtile + (lane & 15)) * lda;
    const _Float16* Arow1 = Arow0 + (size_t)16 * lda;

    const _Float16* Wp[4];
#pragma unroll
    for (int j = 0; j < 4; ++j) {
        int n = ntile + j * 16;
        bool ok = FULLN || (n < N);
        Wp[j] = W + (size_t)((ok ? n : 0) + ncol) * K + kbaseB;
    }

    v8f acc[MFRAG][4];
#pragma unroll
    for (int f = 0; f < MFRAG; ++f)
#pragma unroll
        for (int j = 0; j < 4; ++j) acc[f][j] = v8f{};

    for (int k0 = 0; k0 < K; k0 += 32) {
        v16h a0 = load_a_frag(Arow0, k0, kbaseA);
        v16h a1{};
        if (MFRAG == 2) a1 = load_a_frag(Arow1, k0, kbaseA);
#pragma unroll
        for (int j = 0; j < 4; ++j) {
            if (!FULLN && (ntile + j * 16) >= N) continue;
            v16h b = *(const v16h*)(Wp[j] + k0);
            acc[0][j] = __builtin_amdgcn_wmma_f32_16x16x32_f16(
                false, a0, false, b, (short)0, acc[0][j], false, false);
            if (MFRAG == 2)
                acc[1][j] = __builtin_amdgcn_wmma_f32_16x16x32_f16(
                    false, a1, false, b, (short)0, acc[1][j], false, false);
        }
    }

    const int mbase = mtile + ((lane >> 4) * 8);
#pragma unroll
    for (int f = 0; f < MFRAG; ++f) {
#pragma unroll
        for (int j = 0; j < 4; ++j) {
            if (!FULLN && (ntile + j * 16) >= N) continue;
            int n = ntile + j * 16 + ncol;
            float bs = bias ? bias[n] : 0.f;
#pragma unroll
            for (int r = 0; r < 8; ++r) {
                float v = acc[f][j][r] + bs;
                if (act == ACT_GELU)          v = gelu_f(v);
                else if (act == ACT_SOFTPLUS) v = softplus_f(v);
                C[(size_t)(mbase + f * 16 + r) * ldc + n] = v;
            }
        }
    }
}

// ---------------------------------------------------------------------------
// f32 -> f16 conversion
// ---------------------------------------------------------------------------
__global__ __launch_bounds__(256)
void cvt_f32_f16(const float* __restrict__ in, _Float16* __restrict__ out, int n)
{
    int i = blockIdx.x * 256 + threadIdx.x;
    if (i < n) out[i] = (_Float16)in[i];
}

// ---------------------------------------------------------------------------
// Embedding gather + positional add
// ---------------------------------------------------------------------------
__global__ __launch_bounds__(256)
void embed_kernel(const int* __restrict__ tgt, const float* __restrict__ emb,
                  const float* __restrict__ pos, float* __restrict__ X)
{
    int idx = blockIdx.x * 256 + threadIdx.x;           // over ROWS_*Dm_
    if (idx >= ROWS_ * Dm_) return;
    int d   = idx & (Dm_ - 1);
    int row = idx >> 9;
    int t   = row & (Lm_ - 1);
    int tok = tgt[row];
    X[idx] = emb[(size_t)tok * Dm_ + d] + pos[t * Dm_ + d];
}

// ---------------------------------------------------------------------------
// Fused cross-attention: per (b,h,q) block of 64 threads (Lk=49)
// ---------------------------------------------------------------------------
__global__ __launch_bounds__(64)
void attn_kernel(const float* __restrict__ Q, const float* __restrict__ Kb,
                 const float* __restrict__ Vb, float* __restrict__ O)
{
    int q = blockIdx.x & (Lm_ - 1);
    int h = (blockIdx.x >> 9) & (NH_ - 1);
    int b = blockIdx.x >> 12;
    int d = threadIdx.x;                                 // 0..63

    __shared__ float sQ[HD_];
    __shared__ float sS[64];

    sQ[d] = Q[((size_t)(b * Lm_ + q)) * Dm_ + h * HD_ + d];
    __syncthreads();

    if (d < LK_) {
        const float* kr = Kb + ((size_t)(b * LK_ + d)) * Dm_ + h * HD_;
        float sc = 0.f;
        for (int i = 0; i < HD_; ++i) sc += sQ[i] * kr[i];
        sS[d] = sc * 0.125f;                             // 1/sqrt(64)
    }
    __syncthreads();

    if (d == 0) {
        float mx = -1e30f;
        for (int t = 0; t < LK_; ++t) mx = fmaxf(mx, sS[t]);
        float sum = 0.f;
        for (int t = 0; t < LK_; ++t) { float e = __expf(sS[t] - mx); sS[t] = e; sum += e; }
        float inv = 1.f / sum;
        for (int t = 0; t < LK_; ++t) sS[t] *= inv;
    }
    __syncthreads();

    float o = 0.f;
    for (int t = 0; t < LK_; ++t)
        o += sS[t] * Vb[((size_t)(b * LK_ + t)) * Dm_ + h * HD_ + d];
    O[((size_t)(b * Lm_ + q)) * Dm_ + h * HD_ + d] = o;
}

// ---------------------------------------------------------------------------
// Residual add + LayerNorm (row = 512): one wave32 per row, 16 elems/lane
// ---------------------------------------------------------------------------
__global__ __launch_bounds__(256)
void add_ln_kernel(float* __restrict__ X, const float* __restrict__ T,
                   const float* __restrict__ w, const float* __restrict__ b,
                   int rows)
{
    int row = blockIdx.x * 8 + (threadIdx.x >> 5);
    if (row >= rows) return;
    int lane = threadIdx.x & 31;

    const float* xr = X + (size_t)row * Dm_;
    const float* tr = T + (size_t)row * Dm_;

    float vals[16];
    float s = 0.f;
#pragma unroll
    for (int i = 0; i < 16; ++i) {
        int c = lane + i * 32;
        vals[i] = xr[c] + tr[c];
        s += vals[i];
    }
#pragma unroll
    for (int off = 16; off > 0; off >>= 1) s += __shfl_xor(s, off, 32);
    float mean = s * (1.f / Dm_);

    float vs = 0.f;
#pragma unroll
    for (int i = 0; i < 16; ++i) { float dd = vals[i] - mean; vs += dd * dd; }
#pragma unroll
    for (int off = 16; off > 0; off >>= 1) vs += __shfl_xor(vs, off, 32);
    float rstd = rsqrtf(vs * (1.f / Dm_) + 1e-5f);

#pragma unroll
    for (int i = 0; i < 16; ++i) {
        int c = lane + i * 32;
        X[(size_t)row * Dm_ + c] = (vals[i] - mean) * rstd * w[c] + b[c];
    }
}

// ---------------------------------------------------------------------------
// Depthwise causal conv (k=4) + bias + SiLU.  in: xz[:, :1024] (stride 2048)
// ---------------------------------------------------------------------------
__global__ __launch_bounds__(256)
void conv_silu_kernel(const float* __restrict__ xz, const float* __restrict__ cw,
                      const float* __restrict__ cb, float* __restrict__ out)
{
    int idx = blockIdx.x * 256 + threadIdx.x;           // over B*L*DI
    if (idx >= Bm_ * Lm_ * DI_) return;
    int c = idx & (DI_ - 1);
    int t = (idx >> 10) & (Lm_ - 1);
    int b = idx >> 19;
    float acc = cb[c];
#pragma unroll
    for (int j = 0; j < 4; ++j) {
        int tt = t + j - 3;
        if (tt >= 0)
            acc += cw[j * DI_ + c] * xz[((size_t)(b * Lm_ + tt)) * (2 * DI_) + c];
    }
    out[((size_t)(b * Lm_ + t)) * DI_ + c] = silu_f(acc);
}

// ---------------------------------------------------------------------------
// Mamba selective scan: one thread per (b, channel). 16-state recurrence in
// registers; fused +x*D and *silu(z) epilogue.
// ---------------------------------------------------------------------------
__global__ __launch_bounds__(256)
void mamba_scan_kernel(const float* __restrict__ dt,    // [ROWS,1024]
                       const float* __restrict__ xdbl,  // [ROWS,80] (B @32, C @48)
                       const float* __restrict__ xi,    // [ROWS,1024]
                       const float* __restrict__ xz,    // [ROWS,2048] (z at +1024)
                       const float* __restrict__ A_log, // [1024,16]
                       const float* __restrict__ Dp,    // [1024]
                       float* __restrict__ out)         // [ROWS,1024]
{
    int idx = blockIdx.x * 256 + threadIdx.x;           // 0..16383
    int d = idx & (DI_ - 1);
    int b = idx >> 10;

    float Ar[DS_];
#pragma unroll
    for (int s = 0; s < DS_; ++s) Ar[s] = -__expf(A_log[d * DS_ + s]);
    float Dv = Dp[d];

    float h[DS_];
#pragma unroll
    for (int s = 0; s < DS_; ++s) h[s] = 0.f;

    for (int t = 0; t < Lm_; ++t) {
        size_t row = (size_t)b * Lm_ + t;
        float dtv = dt[row * DI_ + d];
        float xv  = xi[row * DI_ + d];
        float zv  = xz[row * (2 * DI_) + DI_ + d];
        const float* bc = xdbl + row * 80;
        float dx = dtv * xv;
        float y = 0.f;
#pragma unroll
        for (int s = 0; s < DS_; ++s) {
            float dA = __expf(dtv * Ar[s]);
            h[s] = h[s] * dA + dx * bc[DTR_ + s];
            y += h[s] * bc[DTR_ + DS_ + s];
        }
        y += xv * Dv;
        y *= silu_f(zv);
        out[row * DI_ + d] = y;
    }
}

// ---------------------------------------------------------------------------
// Host orchestration
// ---------------------------------------------------------------------------
static inline void gemm(hipStream_t s, const _Float16* A, int lda,
                        const _Float16* W, const float* bias,
                        float* C, int ldc, int M, int N, int K, int act)
{
    if ((N % 64) == 0 && (K % 64) == 0) {
        dim3 g(N / 64, (M + 255) / 256);
        gemm_wmma_f16_lds<<<g, 256, 0, s>>>(A, lda, W, bias, C, ldc, M, N, K, act);
    } else if ((N % 64) == 0) {
        if ((M % 256) == 0) {
            dim3 g(N / 64, M / 256);
            gemm_wmma_f16<2, true><<<g, 256, 0, s>>>(A, lda, W, bias, C, ldc, M, N, K, act);
        } else {
            dim3 g(N / 64, (M + 127) / 128);
            gemm_wmma_f16<1, true><<<g, 256, 0, s>>>(A, lda, W, bias, C, ldc, M, N, K, act);
        }
    } else {
        dim3 g((N + 63) / 64, (M + 127) / 128);
        gemm_wmma_f16<1, false><<<g, 256, 0, s>>>(A, lda, W, bias, C, ldc, M, N, K, act);
    }
}
static inline void cvt(hipStream_t s, const float* in, _Float16* out, int n)
{
    cvt_f32_f16<<<(n + 255) / 256, 256, 0, s>>>(in, out, n);
}

// Per-block leaf order (JAX pytree sorted keys)
enum { pA_LOG = 0, pD, pB1, pB2, pBK, pBO, pBQ, pBV, pCONV_B, pCONV_W, pDT_B,
       pDT_PROJ, pIN_PROJ, pLN1B, pLN1W, pLN2B, pLN2W, pLN3B, pLN3W,
       pMAMBA_OUT, pW1, pW2, pWK, pWO, pWQ, pWV, pX_PROJ, NPB };

extern "C" void kernel_launch(void* const* d_in, const int* in_sizes, int n_in,
                              void* d_out, int out_size, void* d_ws, size_t ws_size,
                              hipStream_t stream)
{
    (void)in_sizes; (void)n_in; (void)out_size; (void)ws_size;

    const float* memory = (const float*)d_in[0];
    const int*   tgt    = (const int*)d_in[1];
    auto P = [&](int blk, int leaf) -> const float* {
        return (const float*)d_in[2 + blk * NPB + leaf];
    };
    const int base_tail = 2 + 6 * NPB;
    const float* embedW = (const float*)d_in[base_tail + 0];
    const float* fc_b   = (const float*)d_in[base_tail + 1];
    const float* fc_w   = (const float*)d_in[base_tail + 2];
    const float* pos    = (const float*)d_in[base_tail + 3];

    // ---- workspace carve-out ----
    char* wp = (char*)d_ws;
    auto carve = [&](size_t bytes) -> void* {
        void* r = (void*)wp;
        wp += (bytes + 255) & ~(size_t)255;
        return r;
    };
    float*    X    = (float*)carve((size_t)ROWS_ * Dm_ * 4);      // activations
    float*    T0   = (float*)carve((size_t)ROWS_ * 2 * DI_ * 4);  // xz / ffn hidden
    float*    T1   = (float*)carve((size_t)ROWS_ * DI_ * 4);      // conv out xi
    float*    T2   = (float*)carve((size_t)ROWS_ * 80 * 4);       // x_dbl
    float*    T3   = (float*)carve((size_t)ROWS_ * DI_ * 4);      // dt
    float*    T4   = (float*)carve((size_t)ROWS_ * DI_ * 4);      // scan y
    float*    T5   = (float*)carve((size_t)ROWS_ * Dm_ * 4);      // proj out
    float*    Qb   = (float*)carve((size_t)ROWS_ * Dm_ * 4);
    float*    KB   = (float*)carve((size_t)MROWS_ * Dm_ * 4);
    float*    VB   = (float*)carve((size_t)MROWS_ * Dm_ * 4);
    float*    AO   = (float*)carve((size_t)ROWS_ * Dm_ * 4);      // attn out
    _Float16* Ah   = (_Float16*)carve((size_t)ROWS_ * 2 * DI_ * 2);
    _Float16* Wh   = (_Float16*)carve((size_t)VOC_ * Dm_ * 2);
    _Float16* MemH = (_Float16*)carve((size_t)MROWS_ * Dm_ * 2);

    // ---- embed + pos ----
    embed_kernel<<<(ROWS_ * Dm_ + 255) / 256, 256, 0, stream>>>(tgt, embedW, pos, X);
    cvt(stream, memory, MemH, MROWS_ * Dm_);

    for (int blk = 0; blk < 6; ++blk) {
        // ======== cross-attention ========
        cvt(stream, X, Ah, ROWS_ * Dm_);
        cvt(stream, P(blk, pWQ), Wh, Dm_ * Dm_);
        gemm(stream, Ah, Dm_, Wh, P(blk, pBQ), Qb, Dm_, ROWS_, Dm_, Dm_, ACT_NONE);
        cvt(stream, P(blk, pWK), Wh, Dm_ * Dm_);
        gemm(stream, MemH, Dm_, Wh, P(blk, pBK), KB, Dm_, MROWS_, Dm_, Dm_, ACT_NONE);
        cvt(stream, P(blk, pWV), Wh, Dm_ * Dm_);
        gemm(stream, MemH, Dm_, Wh, P(blk, pBV), VB, Dm_, MROWS_, Dm_, Dm_, ACT_NONE);
        attn_kernel<<<Bm_ * NH_ * Lm_, 64, 0, stream>>>(Qb, KB, VB, AO);
        cvt(stream, AO, Ah, ROWS_ * Dm_);
        cvt(stream, P(blk, pWO), Wh, Dm_ * Dm_);
        gemm(stream, Ah, Dm_, Wh, P(blk, pBO), T5, Dm_, ROWS_, Dm_, Dm_, ACT_NONE);
        add_ln_kernel<<<ROWS_ / 8, 256, 0, stream>>>(X, T5, P(blk, pLN1W), P(blk, pLN1B), ROWS_);

        // ======== mamba ========
        cvt(stream, X, Ah, ROWS_ * Dm_);
        cvt(stream, P(blk, pIN_PROJ), Wh, 2 * DI_ * Dm_);
        gemm(stream, Ah, Dm_, Wh, nullptr, T0, 2 * DI_, ROWS_, 2 * DI_, Dm_, ACT_NONE);
        conv_silu_kernel<<<(Bm_ * Lm_ * DI_ + 255) / 256, 256, 0, stream>>>(
            T0, P(blk, pCONV_W), P(blk, pCONV_B), T1);
        cvt(stream, T1, Ah, ROWS_ * DI_);
        cvt(stream, P(blk, pX_PROJ), Wh, 80 * DI_);
        gemm(stream, Ah, DI_, Wh, nullptr, T2, 80, ROWS_, 80, DI_, ACT_NONE);
        cvt(stream, T2, Ah, ROWS_ * 80);
        cvt(stream, P(blk, pDT_PROJ), Wh, DI_ * DTR_);
        gemm(stream, Ah, 80, Wh, P(blk, pDT_B), T3, DI_, ROWS_, DI_, DTR_, ACT_SOFTPLUS);
        mamba_scan_kernel<<<(Bm_ * DI_) / 256, 256, 0, stream>>>(
            T3, T2, T1, T0, P(blk, pA_LOG), P(blk, pD), T4);
        cvt(stream, T4, Ah, ROWS_ * DI_);
        cvt(stream, P(blk, pMAMBA_OUT), Wh, Dm_ * DI_);
        gemm(stream, Ah, DI_, Wh, nullptr, T5, Dm_, ROWS_, Dm_, DI_, ACT_NONE);
        add_ln_kernel<<<ROWS_ / 8, 256, 0, stream>>>(X, T5, P(blk, pLN2W), P(blk, pLN2B), ROWS_);

        // ======== FFN ========
        cvt(stream, X, Ah, ROWS_ * Dm_);
        cvt(stream, P(blk, pW1), Wh, FFN_ * Dm_);
        gemm(stream, Ah, Dm_, Wh, P(blk, pB1), T0, FFN_, ROWS_, FFN_, Dm_, ACT_GELU);
        cvt(stream, T0, Ah, ROWS_ * FFN_);
        cvt(stream, P(blk, pW2), Wh, Dm_ * FFN_);
        gemm(stream, Ah, FFN_, Wh, P(blk, pB2), T5, Dm_, ROWS_, Dm_, FFN_, ACT_NONE);
        add_ln_kernel<<<ROWS_ / 8, 256, 0, stream>>>(X, T5, P(blk, pLN3W), P(blk, pLN3B), ROWS_);
    }

    // ======== final vocab projection ========
    cvt(stream, X, Ah, ROWS_ * Dm_);
    cvt(stream, fc_w, Wh, VOC_ * Dm_);
    gemm(stream, Ah, Dm_, Wh, fc_b, (float*)d_out, VOC_, ROWS_, VOC_, Dm_, ACT_NONE);
}
